// DeformableWindowAttention_53738630807591
// MI455X (gfx1250) — matrix operand order repaired and checked by reading, not compiled, not run on hardware
//
#include <hip/hip_runtime.h>
#include <stdint.h>

// ---------------------------------------------------------------------------
// Problem constants (from reference): B=2, DIM=192, H=W=256, HEADS=6, HD=32,
// WS=AWS=8 -> no padding, wnh=wnw=32, 2048 windows, 64 queries/keys per window.
// ---------------------------------------------------------------------------
#define NB    2
#define NDIM  192
#define NPIX  65536      // 256*256
#define NHEAD 6
#define NHD   32

typedef __attribute__((ext_vector_type(16))) __bf16          bf16x16;
typedef __attribute__((ext_vector_type(16))) unsigned short  ushx16;
typedef __attribute__((ext_vector_type(8)))  float           floatx8;
typedef __attribute__((ext_vector_type(4)))  unsigned int    uintx4;
typedef __attribute__((ext_vector_type(8)))  int             intx8;
typedef __attribute__((ext_vector_type(4)))  int             intx4;

union FragAB { ushx16 u; bf16x16 b; };

__device__ __forceinline__ unsigned short f2bf(float f) {
  union { float f; unsigned u; } v; v.f = f;
  unsigned r = v.u + 0x7FFFu + ((v.u >> 16) & 1u);   // round-to-nearest-even
  return (unsigned short)(r >> 16);
}
__device__ __forceinline__ float bf2f(unsigned short h) {
  union { unsigned u; float f; } v; v.u = ((unsigned)h) << 16;
  return v.f;
}

// ---------------------------------------------------------------------------
// Tensor Data Mover: async tiled bf16 copy global -> LDS.
// D# layout per CDNA5 ISA (group0/1 + dim2 in group2).
// 6-arg builtin form (amdgpu-toolchain clang-23); trailing int32x8 zero-filled.
// Issued per-wave (EXEC ignored); completion via s_wait_tensorcnt.
// ---------------------------------------------------------------------------
__device__ __forceinline__ void tdm_load_bf16(
    unsigned lds_off, const void* gaddr,
    unsigned tensor_d0, unsigned tensor_d1, unsigned tensor_d2,
    unsigned tile_d0, unsigned tile_d1, unsigned tile_d2,
    unsigned long long stride0, unsigned long long stride1)
{
  const unsigned long long ga = (unsigned long long)(uintptr_t)gaddr;
  uintx4 g0;
  g0[0] = 1u;                                              // count=1 (valid), load
  g0[1] = lds_off;                                         // lds_addr [63:32]
  g0[2] = (unsigned)(ga & 0xFFFFFFFFull);                  // global_addr lo
  g0[3] = (unsigned)((ga >> 32) & 0x01FFFFFFull)           // global_addr [56:32]
        | (2u << 30);                                      // type=2 ("image")
  intx8 g1;
  g1[0] = (int)(1u << 16);                                 // data_size=1 -> 2 bytes
  g1[1] = (int)((tensor_d0 & 0xFFFFu) << 16);              // tensor_dim0 lo16 @[63:48]
  g1[2] = (int)(((tensor_d0 >> 16) & 0xFFFFu)              // tensor_dim0 hi16
        | ((tensor_d1 & 0xFFFFu) << 16));                  // tensor_dim1 lo16
  g1[3] = (int)(((tensor_d1 >> 16) & 0xFFFFu)              // tensor_dim1 hi16
        | ((tile_d0 & 0xFFFFu) << 16));                    // tile_dim0
  g1[4] = (int)((tile_d1 & 0xFFFFu)                        // tile_dim1
        | ((tile_d2 & 0xFFFFu) << 16));                    // tile_dim2 (0 = 2D)
  g1[5] = (int)(stride0 & 0xFFFFFFFFull);                  // dim0_stride lo32
  g1[6] = (int)(((stride0 >> 32) & 0xFFFFull)              // dim0_stride hi16
        | ((stride1 & 0xFFFFull) << 16));                  // dim1_stride lo16
  g1[7] = (int)((stride1 >> 16) & 0xFFFFFFFFull);          // dim1_stride hi32
  intx4 g2;
  g2[0] = (int)tensor_d2;                                  // tensor_dim2
  g2[1] = 0; g2[2] = 0; g2[3] = 0;                         // no dim3 / no iterate
  intx4 g3; g3[0] = 0; g3[1] = 0; g3[2] = 0; g3[3] = 0;
  intx8 g4;                                                // extra group (zeroed)
  g4[0] = 0; g4[1] = 0; g4[2] = 0; g4[3] = 0;
  g4[4] = 0; g4[5] = 0; g4[6] = 0; g4[7] = 0;
  __builtin_amdgcn_tensor_load_to_lds(g0, g1, g2, g3, g4, 0);
}
__device__ __forceinline__ unsigned lds_offset_of(const void* p) {
  return (unsigned)(uintptr_t)p;   // flat->LDS: hardware uses addr[31:0]
}

// ---------------------------------------------------------------------------
// Kernel A: window pooling -> leaky -> offsets/scales (12-row GEMVs) ->
// per-window sampling coordinates.  coords layout: [bh(12)][wy][wx][pt(64)][2]
// ---------------------------------------------------------------------------
__global__ __launch_bounds__(256)
void coords_kernel(const float* __restrict__ x,
                   const float* __restrict__ off_w, const float* __restrict__ off_b,
                   const float* __restrict__ sc_w,  const float* __restrict__ sc_b,
                   float* __restrict__ coords)
{
  __shared__ float pooled[NDIM];
  __shared__ float offv[12];
  __shared__ float scv[12];
  const int blk = blockIdx.x;              // b*1024 + wy*32 + wx
  const int b = blk >> 10, wy = (blk >> 5) & 31, wx = blk & 31;
  const int t = threadIdx.x;

  if (t < NDIM) {
    const float* base = x + (((size_t)(b * NDIM + t)) << 16) + (wy * 8) * 256 + wx * 8;
    float s = 0.f;
    #pragma unroll
    for (int i = 0; i < 8; i++)
      #pragma unroll
      for (int j = 0; j < 8; j++) s += base[i * 256 + j];
    s *= (1.f / 64.f);
    pooled[t] = (s >= 0.f) ? s : 0.01f * s;   // leaky relu
  }
  __syncthreads();

  if (t < 24) {
    const float* wrow = (t < 12) ? (off_w + t * NDIM) : (sc_w + (t - 12) * NDIM);
    float s = (t < 12) ? off_b[t] : sc_b[t - 12];
    for (int c = 0; c < NDIM; c++) s += wrow[c] * pooled[c];
    if (t < 12) offv[t] = s * (1.f / 32.f);   // / [wnw, wnh] (both 32)
    else        scv[t - 12] = s;
  }
  __syncthreads();

  for (int idx = t; idx < 768; idx += 256) {
    const int h = idx >> 7, rem = idx & 127, pt = rem >> 1, comp = rem & 1;
    const int ay = pt >> 3, ax = pt & 7;
    const float sc = scv[h * 2 + comp], of = offv[h * 2 + comp];
    float basev, winv;
    if (comp == 0) { basev = -1.f + (2.f / 255.f) * (wx * 8 + ax); winv = (ax - 3.5f) * (2.f / 255.f); }
    else           { basev = -1.f + (2.f / 255.f) * (wy * 8 + ay); winv = (ay - 3.5f) * (2.f / 255.f); }
    const int bh = b * NHEAD + h;
    coords[(((size_t)bh * 1024 + wy * 32 + wx) * 64 + pt) * 2 + comp] = basev + winv * sc + of;
  }
}

// ---------------------------------------------------------------------------
// Kernel B/D: 1x1-conv GEMM  Out[m, p] = sum_c W[m, c] * X[c, p] + bias[m]
// Block tile: 64 (M) x 128 (N); K = 192 in 6 steps of 32; 8 waves arranged
// 2 (M) x 4 (N), each doing a 2x2 grid of v_wmma_f32_16x16x32_bf16.
// MODE 0: 576-ch qkv from fp32 x -> bf16 (q_pan, k, v) channel planes
// MODE 1: 192-ch q from fp32 lms -> bf16 planes
// MODE 2: 192-ch proj from bf16 attn planes -> fp32; activation tile is
//         staged by the Tensor Data Mover (tensor_load_to_lds + TENSORcnt).
// ---------------------------------------------------------------------------
template<int MODE>
__global__ __launch_bounds__(256)
void gemm1x1(const float* __restrict__ Xf,
             const unsigned short* __restrict__ Xb,
             const float* __restrict__ Wmat,
             const float* __restrict__ bias,
             unsigned short* __restrict__ o_qp,
             unsigned short* __restrict__ o_k,
             unsigned short* __restrict__ o_v,
             unsigned short* __restrict__ o_q,
             float* __restrict__ o_f32)
{
  __shared__ unsigned short Wt[64][40];     // 64 out-ch x 32 K (+pad, 16B rows)
  __shared__ unsigned short Xt[32][128];    // 32 K x 128 px (TDM-compatible)

  const int tid  = threadIdx.x;
  const int lane = tid & 31;
  const int wave = tid >> 5;
  const int mw = wave >> 2, nw = wave & 3;
  const int b  = blockIdx.z;
  const int m0 = blockIdx.y * 64;
  const int p0 = blockIdx.x * 128;
  const int half = lane >> 4, l16 = lane & 15;

  floatx8 acc[2][2];
  #pragma unroll
  for (int mt = 0; mt < 2; mt++) {
    const int chbase = m0 + mw * 32 + mt * 16 + half * 8;
    #pragma unroll
    for (int nt = 0; nt < 2; nt++)
      #pragma unroll
      for (int r = 0; r < 8; r++) acc[mt][nt][r] = bias[chbase + r];
  }

  const size_t xbase = (size_t)b * NDIM * NPIX + p0;

  for (int k0 = 0; k0 < NDIM; k0 += 32) {
    // ---- activation tile: TDM for bf16 input, manual convert for fp32 ----
    if (MODE == 2) {
      if (wave == 0) {   // one wave issues the async tensor DMA
        tdm_load_bf16(lds_offset_of(&Xt[0][0]), Xb + xbase + (size_t)k0 * NPIX,
                      /*tensor*/ (unsigned)NPIX, (unsigned)NDIM, 1u,
                      /*tile  */ 128u, 32u, 0u,
                      /*stride*/ (unsigned long long)NPIX, 0ull);
      }
    } else {
      for (int i = tid; i < 32 * 128; i += 256) {
        const int r = i >> 7, c = i & 127;
        Xt[r][c] = f2bf(Xf[xbase + (size_t)(k0 + r) * NPIX + c]);
      }
      if (k0 + 32 < NDIM) {  // prefetch next K-slab (global_prefetch_b8)
        __builtin_prefetch(&Xf[xbase + (size_t)(k0 + 32 + (tid >> 3)) * NPIX + ((tid & 7) << 4)], 0, 1);
      }
    }
    // ---- weight tile (fp32 -> bf16), overlaps with the TDM transfer ------
    for (int i = tid; i < 64 * 32; i += 256) {
      const int r = i >> 5, c = i & 31;
      Wt[r][c] = f2bf(Wmat[(m0 + r) * NDIM + k0 + c]);
    }
    if (MODE == 2) __builtin_amdgcn_s_wait_tensorcnt(0);
    __syncthreads();

    // A fragments: 16x32 bf16 per M-tile (contiguous k = e + 16*half, same as B)
    FragAB af[2], bfr[2];
    #pragma unroll
    for (int mt = 0; mt < 2; mt++) {
      const int row = mw * 32 + mt * 16 + l16;
      #pragma unroll
      for (int e = 0; e < 16; e++) {
        const int kc = e + (half << 4);
        af[mt].u[e] = Wt[row][kc];
      }
    }
    // B fragments: 32x16 bf16 per N-tile (K striped over lane halves)
    #pragma unroll
    for (int nt = 0; nt < 2; nt++) {
      const int col = nw * 32 + nt * 16 + l16;
      #pragma unroll
      for (int e = 0; e < 16; e++)
        bfr[nt].u[e] = Xt[e + (half << 4)][col];
    }
    #pragma unroll
    for (int mt = 0; mt < 2; mt++)
      #pragma unroll
      for (int nt = 0; nt < 2; nt++)
        acc[mt][nt] = __builtin_amdgcn_wmma_f32_16x16x32_bf16(
            false, af[mt].b, false, bfr[nt].b, (short)0, acc[mt][nt], false, false);
    __syncthreads();
  }

  // writeback per documented C/D layout: VGPR r -> M = r + half*8, N = l16
  #pragma unroll
  for (int mt = 0; mt < 2; mt++) {
    #pragma unroll
    for (int nt = 0; nt < 2; nt++) {
      const int pix = p0 + nw * 32 + nt * 16 + l16;
      #pragma unroll
      for (int r = 0; r < 8; r++) {
        const int ch = m0 + mw * 32 + mt * 16 + half * 8 + r;
        const float v = acc[mt][nt][r];
        if (MODE == 0) {
          const int s = ch / NDIM, c192 = ch % NDIM;       // (q_pan, k, v)
          unsigned short* dst = (s == 0) ? o_qp : (s == 1) ? o_k : o_v;
          dst[(((size_t)(b * NDIM + c192)) << 16) + pix] = f2bf(v);
        } else if (MODE == 1) {
          o_q[(((size_t)(b * NDIM + ch)) << 16) + pix] = f2bf(v);
        } else {
          o_f32[(((size_t)(b * NDIM + ch)) << 16) + pix] = v;
        }
      }
    }
  }
}

// ---------------------------------------------------------------------------
// Kernel C: per-(window, head) deformable attention.
// Block = 64 threads = 2 waves: wave0 samples K + runs lms-q path (-> ao),
// wave1 samples V + runs pan-q path (-> aop).  Each wave's Q tile
// (8px x 8rows x 32ch) is fetched by an async 3-D TDM transfer that overlaps
// the bilinear K/V gather; attention itself is all bf16 WMMA.
// ---------------------------------------------------------------------------
__global__ __launch_bounds__(64)
void attn_kernel(const unsigned short* __restrict__ qz,
                 const unsigned short* __restrict__ qp,
                 const unsigned short* __restrict__ kkp,
                 const unsigned short* __restrict__ vvp,
                 const float* __restrict__ coords,
                 const float* __restrict__ rpb_table,
                 unsigned short* __restrict__ ao,
                 unsigned short* __restrict__ aop)
{
  __shared__ unsigned short ksT[32][64];      // K^T: [hd][key]
  __shared__ unsigned short vsL[64][32];      // V:   [key][hd]
  __shared__ unsigned short qT[2][32][64];    // TDM dest: [src][hd][query]
  __shared__ unsigned short wsc[2][64][64];   // per-wave softmax probs
  __shared__ float rpbL[225 * NHEAD];

  const int blk = blockIdx.x;
  const int b = blk >> 10, wy = (blk >> 5) & 31, wx = blk & 31;
  const int h = blockIdx.y;
  const int bh = b * NHEAD + h;
  const int tid = threadIdx.x;
  const int lane = tid & 31;                  // channel for staging
  const int wave = tid >> 5;                  // 0: K/lms-q, 1: V/pan-q
  const int half = lane >> 4, l16 = lane & 15;

  // ---- kick off this wave's Q-tile DMA first (overlaps the gathers) -----
  const unsigned short* qs = wave ? qp : qz;
  const size_t qoff = (((size_t)bh * NHD) << 16) + (size_t)(wy * 8) * 256 + wx * 8;
  tdm_load_bf16(lds_offset_of(&qT[wave][0][0]), qs + qoff,
                /*tensor*/ 256u, 256u, 4096u,
                /*tile  */ 8u, 8u, (unsigned)NHD,
                /*stride*/ 256ull, (unsigned long long)NPIX);

  for (int i = tid; i < 225 * NHEAD; i += 64) rpbL[i] = rpb_table[i];

  // -------- bilinear grid sample (zero padding, align_corners) ----------
  const unsigned short* sm = wave ? vvp : kkp;
  const size_t plane = ((size_t)bh * NHD + lane) << 16;
  const float* cw = coords + ((size_t)bh * 1024 + wy * 32 + wx) * 128;
  for (int p = 0; p < 64; p++) {
    const float gx = (cw[p * 2 + 0] + 1.f) * 127.5f;
    const float gy = (cw[p * 2 + 1] + 1.f) * 127.5f;
    const float x0 = floorf(gx), y0 = floorf(gy);
    const float wx1 = gx - x0, wy1 = gy - y0;
    const float wx0 = 1.f - wx1, wy0 = 1.f - wy1;
    const int ix0 = (int)x0, iy0 = (int)y0, ix1 = ix0 + 1, iy1 = iy0 + 1;
    float a = 0.f;
    if (ix0 >= 0 && ix0 < 256 && iy0 >= 0 && iy0 < 256) a += wx0 * wy0 * bf2f(sm[plane + iy0 * 256 + ix0]);
    if (ix1 >= 0 && ix1 < 256 && iy0 >= 0 && iy0 < 256) a += wx1 * wy0 * bf2f(sm[plane + iy0 * 256 + ix1]);
    if (ix0 >= 0 && ix0 < 256 && iy1 >= 0 && iy1 < 256) a += wx0 * wy1 * bf2f(sm[plane + iy1 * 256 + ix0]);
    if (ix1 >= 0 && ix1 < 256 && iy1 >= 0 && iy1 < 256) a += wx1 * wy1 * bf2f(sm[plane + iy1 * 256 + ix1]);
    const unsigned short bv = f2bf(a);
    if (wave == 0) ksT[lane][p] = bv;
    else           vsL[p][lane] = bv;
  }

  __builtin_amdgcn_s_wait_tensorcnt(0);   // Q tile landed in LDS
  __syncthreads();

  // -------- dots = Q * K^T  (4x4 tiles, K=32: one wmma each) ------------
  FragAB aq[4];
  #pragma unroll
  for (int mt = 0; mt < 4; mt++) {
    const int row = mt * 16 + l16;        // query index
    #pragma unroll
    for (int e = 0; e < 16; e++) {
      const int kc = e + (half << 4);   // hd index
      aq[mt].u[e] = qT[wave][kc][row];
    }
  }
  floatx8 d[4][4];
  #pragma unroll
  for (int nt = 0; nt < 4; nt++) {
    FragAB bk;
    const int col = nt * 16 + l16;
    #pragma unroll
    for (int e = 0; e < 16; e++) bk.u[e] = ksT[e + (half << 4)][col];
    #pragma unroll
    for (int mt = 0; mt < 4; mt++) {
      floatx8 z = {0.f, 0.f, 0.f, 0.f, 0.f, 0.f, 0.f, 0.f};
      d[mt][nt] = __builtin_amdgcn_wmma_f32_16x16x32_bf16(
          false, aq[mt].b, false, bk.b, (short)0, z, false, false);
    }
  }

  // -------- softmax rows (scale + rpb, shuffle reduce over 16 lanes) ----
  const float scale = 0.17677669529663687f;   // 32^-0.5
  #pragma unroll
  for (int mt = 0; mt < 4; mt++) {
    #pragma unroll
    for (int nt = 0; nt < 4; nt++) {
      const int kidx = nt * 16 + l16, ky = kidx >> 3, kx2 = kidx & 7;
      #pragma unroll
      for (int r = 0; r < 8; r++) {
        const int q = mt * 16 + half * 8 + r;
        const int ridx = ((q >> 3) - ky + 7) * 15 + ((q & 7) - kx2 + 7);
        d[mt][nt][r] = d[mt][nt][r] * scale + rpbL[ridx * NHEAD + h];
      }
    }
    #pragma unroll
    for (int r = 0; r < 8; r++) {
      float m = d[mt][0][r];
      #pragma unroll
      for (int nt = 1; nt < 4; nt++) m = fmaxf(m, d[mt][nt][r]);
      #pragma unroll
      for (int o = 1; o < 16; o <<= 1) m = fmaxf(m, __shfl_xor(m, o, 32));
      float s = 0.f, e0[4];
      #pragma unroll
      for (int nt = 0; nt < 4; nt++) { e0[nt] = __expf(d[mt][nt][r] - m); s += e0[nt]; }
      #pragma unroll
      for (int o = 1; o < 16; o <<= 1) s += __shfl_xor(s, o, 32);
      const float inv = 1.f / s;
      const int q = mt * 16 + half * 8 + r;
      #pragma unroll
      for (int nt = 0; nt < 4; nt++)
        wsc[wave][q][nt * 16 + l16] = f2bf(e0[nt] * inv);
    }
  }

  // -------- out = probs * V  (4x2 tiles, K=64 in 2 steps) ---------------
  floatx8 o[4][2];
  #pragma unroll
  for (int mt = 0; mt < 4; mt++)
    #pragma unroll
    for (int nt = 0; nt < 2; nt++) o[mt][nt] = (floatx8){0.f, 0.f, 0.f, 0.f, 0.f, 0.f, 0.f, 0.f};
  #pragma unroll
  for (int k2 = 0; k2 < 2; k2++) {
    FragAB ap[4], bv2[2];
    #pragma unroll
    for (int mt = 0; mt < 4; mt++) {
      const int row = mt * 16 + l16;
      #pragma unroll
      for (int e = 0; e < 16; e++) {
        const int kc = k2 * 32 + e + (half << 4);
        ap[mt].u[e] = wsc[wave][row][kc];
      }
    }
    #pragma unroll
    for (int nt = 0; nt < 2; nt++) {
      const int col = nt * 16 + l16;
      #pragma unroll
      for (int e = 0; e < 16; e++) bv2[nt].u[e] = vsL[k2 * 32 + e + (half << 4)][col];
    }
    #pragma unroll
    for (int mt = 0; mt < 4; mt++)
      #pragma unroll
      for (int nt = 0; nt < 2; nt++)
        o[mt][nt] = __builtin_amdgcn_wmma_f32_16x16x32_bf16(
            false, ap[mt].b, false, bv2[nt].b, (short)0, o[mt][nt], false, false);
  }

  // -------- scatter to channel planes (bf16) ----------------------------
  unsigned short* dst = wave ? aop : ao;
  #pragma unroll
  for (int mt = 0; mt < 4; mt++) {
    #pragma unroll
    for (int nt = 0; nt < 2; nt++) {
      const int hd = nt * 16 + l16;
      const size_t chb = ((size_t)(b * NDIM + h * NHD + hd)) << 16;
      #pragma unroll
      for (int r = 0; r < 8; r++) {
        const int q = mt * 16 + half * 8 + r;
        const int pix = (wy * 8 + (q >> 3)) * 256 + wx * 8 + (q & 7);
        dst[chb + pix] = f2bf(o[mt][nt][r]);
      }
    }
  }
}

// ---------------------------------------------------------------------------
extern "C" void kernel_launch(void* const* d_in, const int* in_sizes, int n_in,
                              void* d_out, int out_size, void* d_ws, size_t ws_size,
                              hipStream_t stream) {
  (void)in_sizes; (void)n_in; (void)out_size; (void)ws_size;
  const float* x      = (const float*)d_in[0];
  const float* lms    = (const float*)d_in[1];
  const float* qkv_w  = (const float*)d_in[2];
  const float* qkv_b  = (const float*)d_in[3];
  const float* off_w  = (const float*)d_in[4];
  const float* off_b  = (const float*)d_in[5];
  const float* sc_w   = (const float*)d_in[6];
  const float* sc_b   = (const float*)d_in[7];
  const float* proj_w = (const float*)d_in[8];
  const float* proj_b = (const float*)d_in[9];
  const float* rpb    = (const float*)d_in[10];

  char* wsp = (char*)d_ws;
  size_t off = 0;
  auto carve = [&](size_t bytes) -> void* {
    void* p = wsp + off;
    off = (off + bytes + 255) & ~(size_t)255;
    return p;
  };
  float* coords = (float*)carve((size_t)12 * 1024 * 64 * 2 * sizeof(float));
  const size_t planeBytes = (size_t)NB * NDIM * NPIX * sizeof(unsigned short);
  unsigned short* qzb  = (unsigned short*)carve(planeBytes);   // q (from lms)
  unsigned short* qpb  = (unsigned short*)carve(planeBytes);   // q_pan (from x)
  unsigned short* kkb  = (unsigned short*)carve(planeBytes);   // k
  unsigned short* vvb  = (unsigned short*)carve(planeBytes);   // v
  unsigned short* aob  = (unsigned short*)carve(planeBytes);   // attn out (lms path)
  unsigned short* aopb = (unsigned short*)carve(planeBytes);   // attn out (pan path)

  coords_kernel<<<dim3(2048), dim3(256), 0, stream>>>(x, off_w, off_b, sc_w, sc_b, coords);

  gemm1x1<0><<<dim3(512, 9, NB), dim3(256), 0, stream>>>(
      x, nullptr, qkv_w, qkv_b, qpb, kkb, vvb, nullptr, nullptr);

  gemm1x1<1><<<dim3(512, 3, NB), dim3(256), 0, stream>>>(
      lms, nullptr, qkv_w, qkv_b, nullptr, nullptr, nullptr, qzb, nullptr);

  attn_kernel<<<dim3(2048, NHEAD), dim3(64), 0, stream>>>(
      qzb, qpb, kkb, vvb, coords, rpb, aob, aopb);

  float* out0 = (float*)d_out;
  float* out1 = out0 + (size_t)NB * NDIM * NPIX;
  gemm1x1<2><<<dim3(512, 3, NB), dim3(256), 0, stream>>>(
      nullptr, aob, proj_w, proj_b, nullptr, nullptr, nullptr, nullptr, out0);
  gemm1x1<2><<<dim3(512, 3, NB), dim3(256), 0, stream>>>(
      nullptr, aopb, proj_w, proj_b, nullptr, nullptr, nullptr, nullptr, out1);
}
